// LatNet_35579509080783
// MI455X (gfx1250) — matrix operand order repaired and checked by reading, not compiled, not run on hardware
//
#include <hip/hip_runtime.h>
#include <hip/hip_bf16.h>
#include <stdint.h>

typedef __bf16 bf16;
typedef __attribute__((ext_vector_type(16))) __bf16 v16bf;
typedef __attribute__((ext_vector_type(8)))  __bf16 v8bf;
typedef __attribute__((ext_vector_type(8)))  float  v8f;

#define N_NODES 32768
#define N_EDGES 262144
#define SLOPE 0.1f
#define BN_EPS 1e-5f

__device__ __forceinline__ float lrelu(float x) { return x >= 0.f ? x : SLOPE * x; }

// Load a 16-element bf16 fragment from two aligned 16-byte halves at p and p+elemOff.
__device__ __forceinline__ v16bf load2x8(const bf16* p, int elemOff) {
    v8bf lo = *(const v8bf*)p;
    v8bf hi = *(const v8bf*)(p + elemOff);
    return __builtin_shufflevector(lo, hi, 0,1,2,3,4,5,6,7,8,9,10,11,12,13,14,15);
}

// ---------------- elementwise / setup kernels ----------------

__global__ void fill_f32(float* p, float v, int n) {
    int i = blockIdx.x * blockDim.x + threadIdx.x;
    if (i < n) p[i] = v;
}

__global__ void deg_accum(const int* __restrict__ dst, const float* __restrict__ ew,
                          float* deg, int E) {
    int e = blockIdx.x * blockDim.x + threadIdx.x;
    if (e < E) atomicAdd(&deg[dst[e]], ew[e]);
}

__global__ void deg_finalize(const float* __restrict__ deg, float* dis, int n) {
    int i = blockIdx.x * blockDim.x + threadIdx.x;
    if (i < n) dis[i] = rsqrtf(deg[i] + 1.0f);
}

// Convert f32 weight W[K,N] -> bf16 packed in WMMA B-fragment order:
// out[((nt*(K/32) + k0)*32 + lane)*16 + v]  where lane = half*16 + (n%16),
// element v corresponds to K index k0*32 + half*8 + v (v<8), k0*32 + 16 + half*8 + (v-8) (v>=8).
__global__ void pack_B_bf16(const float* __restrict__ W, bf16* __restrict__ out,
                            int K, int N) {
    int i = blockIdx.x * blockDim.x + threadIdx.x;
    if (i >= K * N) return;
    int k = i / N, n = i - k * N;
    int k0 = k >> 5, kin = k & 31;
    int half = (kin >> 3) & 1, v = (kin & 7) | ((kin >> 4) << 3);
    int nt = n >> 4, r = n & 15;
    int lane = half * 16 + r;
    size_t o = (((size_t)nt * (K >> 5) + k0) * 32 + (size_t)lane) * 16 + v;
    out[o] = (bf16)W[i];
}

// ---------------- bf16 WMMA GEMM (register double-buffered) ----------------
// C[M,N] = A[M,K](bf16 row-major) * Bp (bf16, fragment-packed) (+bias)(+lrelu)
// One wave computes a 16 x (16*WN) strip. kblocks (=K/32) must be even and >= 2.
template <int WN>
__global__ void wmma_gemm_bf16(const bf16* __restrict__ A, const bf16* __restrict__ Bp,
                               const float* __restrict__ bias,
                               float* __restrict__ Cf, bf16* __restrict__ Cb,
                               int M, int N, int K, int act) {
    int wave    = (int)((blockIdx.x * blockDim.x + threadIdx.x) >> 5);
    int groupsN = (N >> 4) / WN;
    int waves   = (M >> 4) * groupsN;
    if (wave >= waves) return;
    int tm = wave / groupsN, tg = wave - tm * groupsN;
    int m0  = tm << 4;
    int nt0 = tg * WN;

    int lane = threadIdx.x & 31;
    int half = lane >> 4;   // 0: lanes 0-15, 1: lanes 16-31
    int r    = lane & 15;

    v8f acc[WN];
#pragma unroll
    for (int w = 0; w < WN; ++w) acc[w] = (v8f){};

    int kblocks = K >> 5;
    // A fragment: row m0+r, elements 0..7 = K half*8+0..7, 8..15 = K half*8+16..23
    const bf16* Arow = A + (size_t)(m0 + r) * K + half * 8;
    // Packed B: 512 elems per (n-tile, k-block); per-lane 16 contiguous bf16
    const bf16* bbase = Bp + (size_t)nt0 * kblocks * 512 + (size_t)lane * 16;
    const size_t nstride = (size_t)kblocks * 512;

    // ---- software pipeline: preload k-block 0 ----
    v16bf a0, a1, b0[WN], b1[WN];
    a0 = load2x8(Arow, 16);
#pragma unroll
    for (int w = 0; w < WN; ++w) b0[w] = load2x8(bbase + (size_t)w * nstride, 8);
    Arow += 32; bbase += 512;

    for (int kb = 0; kb < kblocks; kb += 2) {
        // issue loads for block kb+1 while block kb sits in registers
        a1 = load2x8(Arow, 16);
#pragma unroll
        for (int w = 0; w < WN; ++w) b1[w] = load2x8(bbase + (size_t)w * nstride, 8);
        Arow += 32; bbase += 512;

#pragma unroll
        for (int w = 0; w < WN; ++w)
            acc[w] = __builtin_amdgcn_wmma_f32_16x16x32_bf16(
                false, a0, false, b0[w], (short)0, acc[w], false, false);

        // issue loads for block kb+2 while block kb+1 is consumed
        if (kb + 2 < kblocks) {
            a0 = load2x8(Arow, 16);
#pragma unroll
            for (int w = 0; w < WN; ++w) b0[w] = load2x8(bbase + (size_t)w * nstride, 8);
        }
        Arow += 32; bbase += 512;

#pragma unroll
        for (int w = 0; w < WN; ++w)
            acc[w] = __builtin_amdgcn_wmma_f32_16x16x32_bf16(
                false, a1, false, b1[w], (short)0, acc[w], false, false);
    }

#pragma unroll
    for (int w = 0; w < WN; ++w) {
        int n0 = (nt0 + w) << 4;
        float bv = bias ? bias[n0 + r] : 0.f;
#pragma unroll
        for (int v = 0; v < 8; ++v) {
            int row = m0 + half * 8 + v;   // acc VGPR v -> row m0 + half*8 + v
            float xv = acc[w][v] + bv;
            if (act) xv = lrelu(xv);
            size_t o = (size_t)row * N + n0 + r;
            if (Cf) Cf[o] = xv;
            if (Cb) Cb[o] = (bf16)xv;
        }
    }
}

// ---------------- fallback fp32 GEMM (layer0 K=4, final FC N=1) ----------------
__global__ void simple_gemm(const float* __restrict__ A, const float* __restrict__ B,
                            const float* __restrict__ bias, float* __restrict__ C,
                            int M, int N, int K, int act) {
    int i = blockIdx.x * blockDim.x + threadIdx.x;
    if (i >= M * N) return;
    int m = i / N, n = i - m * N;
    float s = bias ? bias[n] : 0.f;
    for (int k = 0; k < K; ++k) s += A[(size_t)m * K + k] * B[(size_t)k * N + n];
    if (act) s = lrelu(s);
    C[i] = s;
}

// ---------------- GCN scatter ----------------
__global__ void conv_self(const float* __restrict__ hlin, const float* __restrict__ dis,
                          const float* __restrict__ bias, float* __restrict__ out,
                          int n_nodes, int C) {
    size_t i = (size_t)blockIdx.x * blockDim.x + threadIdx.x;
    if (i >= (size_t)n_nodes * C) return;
    int n = (int)(i / C), c = (int)(i - (size_t)n * C);
    float d = dis[n];
    out[i] = hlin[i] * d * d + bias[c];
}

__global__ void conv_scatter(const float* __restrict__ hlin, const int* __restrict__ src,
                             const int* __restrict__ dst, const float* __restrict__ ew,
                             const float* __restrict__ dis, float* out, int E, int C) {
    int cg4 = C >> 2;
    long long i = (long long)blockIdx.x * blockDim.x + threadIdx.x;
    if (i >= (long long)E * cg4) return;
    int e = (int)(i / cg4);
    int c = (int)(i - (long long)e * cg4) << 2;
    int s = src[e], d = dst[e];
    float norm = dis[s] * ew[e] * dis[d];
    const float4 h = *(const float4*)(hlin + (size_t)s * C + c);
    float* o = out + (size_t)d * C + c;
    atomicAdd(o + 0, h.x * norm);
    atomicAdd(o + 1, h.y * norm);
    atomicAdd(o + 2, h.z * norm);
    atomicAdd(o + 3, h.w * norm);
}

// ---------------- BatchNorm over lrelu(h) ----------------
__global__ void bn_stats(const float* __restrict__ h, int n_rows, int C,
                         float* __restrict__ mean, float* __restrict__ rstd) {
    __shared__ float s1[256], s2[256];
    int c = blockIdx.x;
    float a = 0.f, b = 0.f;
    for (int n = threadIdx.x; n < n_rows; n += 256) {
        float x = lrelu(h[(size_t)n * C + c]);
        a += x; b += x * x;
    }
    s1[threadIdx.x] = a; s2[threadIdx.x] = b;
    __syncthreads();
    for (int off = 128; off > 0; off >>= 1) {
        if (threadIdx.x < off) {
            s1[threadIdx.x] += s1[threadIdx.x + off];
            s2[threadIdx.x] += s2[threadIdx.x + off];
        }
        __syncthreads();
    }
    if (threadIdx.x == 0) {
        float m = s1[0] / n_rows;
        float v = s2[0] / n_rows - m * m;
        mean[c] = m;
        rstd[c] = rsqrtf(v + BN_EPS);
    }
}

__global__ void bn_apply(const float* __restrict__ h, const float* __restrict__ mean,
                         const float* __restrict__ rstd, const float* __restrict__ gamma,
                         const float* __restrict__ beta, bf16* __restrict__ out,
                         int n_rows, int C) {
    size_t i = (size_t)blockIdx.x * blockDim.x + threadIdx.x;
    if (i >= (size_t)n_rows * C) return;
    int c = (int)(i % C);
    float x = lrelu(h[i]);
    out[i] = (bf16)(gamma[c] * (x - mean[c]) * rstd[c] + beta[c]);
}

// ---------------- host ----------------

static const int convDims[8][2] = {{4,256},{256,512},{512,1024},{1024,2048},
                                   {2048,1024},{1024,512},{512,256},{256,128}};
static const int fcDims[8][2]   = {{4096,4096},{4096,1024},{1024,512},{512,256},
                                   {256,128},{128,64},{64,32},{32,1}};

static inline int cdiv(long long a, long long b) { return (int)((a + b - 1) / b); }

extern "C" void kernel_launch(void* const* d_in, const int* in_sizes, int n_in,
                              void* d_out, int out_size, void* d_ws, size_t ws_size,
                              hipStream_t stream) {
    const float* x  = (const float*)d_in[0];
    const int*   ei = (const int*)d_in[1];
    const float* ew = (const float*)d_in[2];
    const int* src = ei;
    const int* dst = ei + N_EDGES;
    const float* convW[8]; const float* convB[8];
    const float* bnG[8];   const float* bnB[8];
    const float* fcW[8];   const float* fcB[8];
    for (int i = 0; i < 8; ++i) {
        convW[i] = (const float*)d_in[3 + i];
        convB[i] = (const float*)d_in[11 + i];
        bnG[i]   = (const float*)d_in[19 + i];
        bnB[i]   = (const float*)d_in[27 + i];
        fcW[i]   = (const float*)d_in[35 + i];
        fcB[i]   = (const float*)d_in[43 + i];
    }

    // ---- carve workspace ----
    uintptr_t base = (uintptr_t)d_ws;
    auto carve = [&](size_t bytes) -> void* {
        uintptr_t r = (base + 255) & ~(uintptr_t)255;
        base = r + bytes;
        return (void*)r;
    };
    bf16*  act   = (bf16*) carve((size_t)N_NODES * 2048 * sizeof(bf16));
    float* hlin  = (float*)carve((size_t)N_NODES * 2048 * sizeof(float));
    float* hconv = (float*)carve((size_t)N_NODES * 2048 * sizeof(float));
    float* deg   = (float*)carve(N_NODES * sizeof(float));
    float* dis   = (float*)carve(N_NODES * sizeof(float));
    float* mean  = (float*)carve(2048 * sizeof(float));
    float* rstd  = (float*)carve(2048 * sizeof(float));
    size_t wElems = 0;
    for (int i = 1; i < 8; ++i) wElems += (size_t)convDims[i][0] * convDims[i][1];
    for (int i = 0; i < 7; ++i) wElems += (size_t)fcDims[i][0] * fcDims[i][1];
    bf16* wbf = (bf16*)carve(wElems * sizeof(bf16));
    // FC ping-pong buffers overlay hlin region (hlin is dead during the FC chain)
    bf16*  fcA  = (bf16*)hlin;
    bf16*  fcB2 = fcA + (size_t)1024 * 4096;
    float* fcF  = (float*)(fcB2 + (size_t)1024 * 4096);

    const int T = 256;

    // ---- degree / normalization ----
    fill_f32<<<cdiv(N_NODES, T), T, 0, stream>>>(deg, 0.f, N_NODES);
    deg_accum<<<cdiv(N_EDGES, T), T, 0, stream>>>(dst, ew, deg, N_EDGES);
    deg_finalize<<<cdiv(N_NODES, T), T, 0, stream>>>(deg, dis, N_NODES);

    // ---- convert + fragment-pack weights to bf16 ----
    bf16* wConvBf[8] = {};
    bf16* wFcBf[8]   = {};
    {
        size_t off = 0;
        for (int i = 1; i < 8; ++i) {
            int K = convDims[i][0], N = convDims[i][1];
            size_t n = (size_t)K * N;
            wConvBf[i] = wbf + off;
            pack_B_bf16<<<cdiv((long long)n, T), T, 0, stream>>>(convW[i], wConvBf[i], K, N);
            off += n;
        }
        for (int i = 0; i < 7; ++i) {
            int K = fcDims[i][0], N = fcDims[i][1];
            size_t n = (size_t)K * N;
            wFcBf[i] = wbf + off;
            pack_B_bf16<<<cdiv((long long)n, T), T, 0, stream>>>(fcW[i], wFcBf[i], K, N);
            off += n;
        }
    }

    // ---- conv layers ----
    for (int L = 0; L < 8; ++L) {
        int ci = convDims[L][0], co = convDims[L][1];
        if (L == 0) {
            simple_gemm<<<cdiv((long long)N_NODES * co, T), T, 0, stream>>>(
                x, convW[0], nullptr, hlin, N_NODES, co, ci, 0);
        } else {
            int waves = (N_NODES / 16) * ((co / 16) / 4);
            wmma_gemm_bf16<4><<<cdiv((long long)waves * 32, T), T, 0, stream>>>(
                act, wConvBf[L], nullptr, hlin, nullptr, N_NODES, co, ci, 0);
        }
        conv_self<<<cdiv((long long)N_NODES * co, T), T, 0, stream>>>(
            hlin, dis, convB[L], hconv, N_NODES, co);
        conv_scatter<<<cdiv((long long)N_EDGES * (co / 4), T), T, 0, stream>>>(
            hlin, src, dst, ew, dis, hconv, N_EDGES, co);
        bn_stats<<<co, T, 0, stream>>>(hconv, N_NODES, co, mean, rstd);
        bn_apply<<<cdiv((long long)N_NODES * co, T), T, 0, stream>>>(
            hconv, mean, rstd, bnG[L], bnB[L], act, N_NODES, co);
    }

    // ---- FC chain: act is [1024, 4096] bf16 (reshape is a view) ----
    const bf16* cur = act;
    bf16* nxt = fcA;
    for (int L = 0; L < 7; ++L) {
        int K = fcDims[L][0], N = fcDims[L][1];
        float* f32out = (L == 6) ? fcF : nullptr;  // feeds fp32 fallback GEMM
        if (N >= 64) {
            int waves = (1024 / 16) * ((N / 16) / 4);
            wmma_gemm_bf16<4><<<cdiv((long long)waves * 32, T), T, 0, stream>>>(
                cur, wFcBf[L], fcB[L], f32out, nxt, 1024, N, K, 1);
        } else {  // N == 32
            int waves = (1024 / 16) * ((N / 16) / 2);
            wmma_gemm_bf16<2><<<cdiv((long long)waves * 32, T), T, 0, stream>>>(
                cur, wFcBf[L], fcB[L], f32out, nxt, 1024, N, K, 1);
        }
        cur = nxt;
        nxt = (nxt == fcA) ? fcB2 : fcA;
    }
    // final FC: [1024,32] x [32,1] + b -> d_out
    simple_gemm<<<cdiv(1024, T), T, 0, stream>>>(
        fcF, fcW[7], fcB[7], (float*)d_out, 1024, 1, 32, 0);
}
